// IonisGateV26_18519898980985
// MI455X (gfx1250) — compile-verified, matrix operand order checked by reading, not run on hardware
//
#include <hip/hip_runtime.h>
#include <hip/hip_bf16.h>

// CDNA5 / gfx1250: wave32, WMMA 16x16x32 bf16 (f32 accumulate), async LDS loads.

typedef __attribute__((ext_vector_type(16))) __bf16 v16bf;
typedef __attribute__((ext_vector_type(8)))  __bf16 v8bf;
typedef __attribute__((ext_vector_type(8)))  float  v8f;

// ---- fast activations: 1 transcendental + ~5 VALU each ---------------------
// mish(x) = x * tanh(softplus(x)) = x * (w^2-1)/(w^2+1),  w = 1 + e^x.
// Clamp exp arg so w^2 stays finite; ratio saturates to exactly 1 (x large)
// and 0 (x very negative), matching mish asymptotes in f32.
__device__ __forceinline__ float mish_f(float x) {
  float e  = __expf(fminf(x, 30.0f));
  float w  = 1.0f + e;
  float w2 = w * w;
  return x * (w2 - 1.0f) * __builtin_amdgcn_rcpf(w2 + 1.0f);
}
// tanh(x) = 1 - 2/(e^{2x}+1); exp underflow/overflow saturate correctly.
__device__ __forceinline__ float tanh_f(float x) {
  float e = __expf(fminf(2.0f * x, 60.0f));
  return 1.0f - 2.0f * __builtin_amdgcn_rcpf(e + 1.0f);
}
__device__ __forceinline__ float sigmoid_f(float x) {
  float e = __expf(fminf(-x, 60.0f));
  return __builtin_amdgcn_rcpf(1.0f + e);
}
// Stable softplus (only used in the 16-lane scalar tail on tiny weights).
__device__ __forceinline__ float softplus_f(float x) {
  return fmaxf(x, 0.0f) + __logf(1.0f + __expf(-fabsf(x)));
}

// ---------------------------------------------------------------------------
// Pack a KxN f32 row-major weight matrix (optionally batched) into bf16 WMMA
// B-fragments.  Fragment (kt, nt) = 32 lanes x 16 bf16:
//   lane L: column n = nt*16 + (L & 15), lane-group g = L >> 4
//   element e in [0,8):  K = kt*32 + 8g + e
//   element e in [8,16): K = kt*32 + 16 + 8g + (e-8)
// Mirrors the ISA 16-bit matrix striping; K zero-padded (handles tw1's K=15).
// ---------------------------------------------------------------------------
__global__ void pack_wmma_b(const float* __restrict__ src, __bf16* __restrict__ dst,
                            int K, int N, int nbatch) {
  const int KT = (K + 31) >> 5;
  const int NT = N >> 4;
  const int perBatch = KT * NT * 512;
  const int total = perBatch * nbatch;
  for (int idx = blockIdx.x * blockDim.x + threadIdx.x; idx < total;
       idx += gridDim.x * blockDim.x) {
    int b      = idx / perBatch;
    int r      = idx - b * perBatch;
    int frag   = r >> 9;
    int within = r & 511;
    int lane   = within >> 4;
    int e      = within & 15;
    int kt = frag / NT;
    int nt = frag - kt * NT;
    int g  = lane >> 4;
    int n0 = lane & 15;
    int k  = kt * 32 + ((e < 8) ? (8 * g + e) : (16 + 8 * g + (e - 8)));
    int n  = nt * 16 + n0;
    float v = (k < K) ? src[(size_t)b * K * N + (size_t)k * N + n] : 0.0f;
    dst[idx] = (__bf16)v;
  }
}

// Load a 16x32 bf16 A-fragment from a row-major LDS activation buffer.
template <int STRIDE>
__device__ __forceinline__ v16bf load_afrag(const __bf16* act, int kbase, int r, int g) {
  v8bf lo = *(const v8bf*)(act + r * STRIDE + kbase + 8 * g);
  v8bf hi = *(const v8bf*)(act + r * STRIDE + kbase + 16 + 8 * g);
  v16bf a;
#pragma unroll
  for (int i = 0; i < 8; ++i) { a[i] = lo[i]; a[i + 8] = hi[i]; }
  return a;
}

// ---------------------------------------------------------------------------
// Fused network: one workgroup (8 wave32) per 16-row tile.
// ---------------------------------------------------------------------------
__global__ __launch_bounds__(256)
void ionis_fused(const float* __restrict__ x,
                 const __bf16* __restrict__ W1,   // 32 frags   (15x512)
                 const __bf16* __restrict__ W2,   // 256 frags  (512x256)
                 const __bf16* __restrict__ WH,   // 9*64 frags (256x128 each)
                 const __bf16* __restrict__ WS,   // 32 frags   (256x64)
                 const __bf16* __restrict__ WT,   // 32 frags   (256x64)
                 const float* __restrict__ tb1, const float* __restrict__ tb2,
                 const float* __restrict__ hb1, const float* __restrict__ hb2,
                 const float* __restrict__ hw2,
                 const float* __restrict__ sb1, const float* __restrict__ sb2,
                 const float* __restrict__ sw2,
                 const float* __restrict__ stb1, const float* __restrict__ stb2,
                 const float* __restrict__ stw2,
                 const float* __restrict__ sun_w1, const float* __restrict__ sun_b1,
                 const float* __restrict__ sun_w2, const float* __restrict__ sun_b2,
                 const float* __restrict__ storm_w1, const float* __restrict__ storm_b1,
                 const float* __restrict__ storm_w2, const float* __restrict__ storm_b2,
                 float* __restrict__ out, int B) {
  __shared__ __align__(16) float  xtile[16 * 18];
  __shared__ __align__(16) __bf16 act1[16][512];
  __shared__ __align__(16) __bf16 act2[16][256];
  __shared__ float heads_acc[16][9];
  __shared__ float gate_acc[16][2];

  const int tid  = threadIdx.x;
  const int lane = tid & 31;
  const int w    = tid >> 5;
  const int g    = lane >> 4;
  const int n0   = lane & 15;
  const int rowbase = blockIdx.x * 16;

  // Stage the 16x18 input slab (1152 B, 16B-aligned since 1152%16==0) with the
  // gfx1250 async global->LDS pipe: 72 lanes x b128, tracked by ASYNCcnt.
  if (tid < 72) {
    unsigned lds_off = (unsigned)(uintptr_t)(&xtile[tid * 4]);   // low 32 bits = LDS offset
    const float* gptr = x + (size_t)rowbase * 18 + tid * 4;
    asm volatile("global_load_async_to_lds_b128 %0, %1, off"
                 :: "v"(lds_off), "v"(gptr) : "memory");
  }
  // Zero the fold buffers while the async copy is in flight.
  if (tid < 144) heads_acc[tid / 9][tid % 9] = 0.0f;
  if (tid < 32)  gate_acc[tid >> 1][tid & 1] = 0.0f;
  if (tid == 0)  __builtin_prefetch(x + (size_t)(rowbase + 16) * 18, 0, 1);
  asm volatile("s_wait_asynccnt 0x0" ::: "memory");
  __syncthreads();

  // ---- phase 1: layer1, K=15 zero-padded to one 16x16x32 WMMA step --------
  v16bf afrag1;
  {
    const int r = n0;
#pragma unroll
    for (int e = 0; e < 8; ++e) {
      int k = 8 * g + e;
      afrag1[e]     = (__bf16)((k < 15) ? xtile[r * 18 + k] : 0.0f);
      afrag1[e + 8] = (__bf16)0.0f;  // K = 16..31 is all padding
    }
  }
  for (int t = 0; t < 4; ++t) {
    const int nt = w * 4 + t;
    v8f acc = {};
    v16bf b = *(const v16bf*)(W1 + ((size_t)nt * 32 + lane) * 16);
    acc = __builtin_amdgcn_wmma_f32_16x16x32_bf16(false, afrag1, false, b,
                                                  (short)0, acc, false, false);
    const int col = nt * 16 + n0;
    const float bias = tb1[col];
#pragma unroll
    for (int i = 0; i < 8; ++i)
      act1[i + 8 * g][col] = (__bf16)mish_f(acc[i] + bias);
  }
  __syncthreads();

  // ---- phase 2: layer2, K=512 (16 WMMA K-steps), 2 N-tiles per wave -------
  for (int t = 0; t < 2; ++t) {
    const int nt = w * 2 + t;
    v8f acc = {};
#pragma unroll
    for (int kt = 0; kt < 16; ++kt) {
      v16bf a = load_afrag<512>(&act1[0][0], kt * 32, n0, g);
      v16bf b = *(const v16bf*)(W2 + (((size_t)kt * 16 + nt) * 32 + lane) * 16);
      acc = __builtin_amdgcn_wmma_f32_16x16x32_bf16(false, a, false, b,
                                                    (short)0, acc, false, false);
    }
    const int col = nt * 16 + n0;
    const float bias = tb2[col];
#pragma unroll
    for (int i = 0; i < 8; ++i)
      act2[i + 8 * g][col] = (__bf16)mish_f(acc[i] + bias);
  }
  __syncthreads();

  // ---- phase 3: 72 head tiles (9 heads x 8 N-tiles) + 8 gate tiles --------
  for (int t = 0; t < 10; ++t) {
    const int tile = w * 10 + t;            // wave-uniform: EXEC stays all-1s
    const bool is_head = (tile < 72);
    int hk = 0, nt, gsel = 0;
    if (is_head) { hk = tile >> 3; nt = tile & 7; }
    else         { int gd = tile - 72; gsel = gd >> 2; nt = gd & 3; }

    v8f acc = {};
#pragma unroll
    for (int kt = 0; kt < 8; ++kt) {
      v16bf a = load_afrag<256>(&act2[0][0], kt * 32, n0, g);
      const __bf16* wp;
      size_t fidx;
      if (is_head)      { wp = WH; fidx = (size_t)hk * 64 + (size_t)kt * 8 + nt; }
      else if (!gsel)   { wp = WS; fidx = (size_t)kt * 4 + nt; }
      else              { wp = WT; fidx = (size_t)kt * 4 + nt; }
      v16bf b = *(const v16bf*)(wp + (fidx * 32 + lane) * 16);
      acc = __builtin_amdgcn_wmma_f32_16x16x32_bf16(false, a, false, b,
                                                    (short)0, acc, false, false);
    }
    const int col = nt * 16 + n0;
    if (is_head) {
      const float bias = hb1[hk * 128 + col];
      const float wv   = hw2[hk * 128 + col];
#pragma unroll
      for (int i = 0; i < 8; ++i) {
        float hv = mish_f(acc[i] + bias);
        atomicAdd(&heads_acc[i + 8 * g][hk], hv * wv);   // ds_add_f32
      }
    } else {
      const float bias = (gsel ? stb1 : sb1)[col];
      const float wv   = (gsel ? stw2 : sw2)[col];
#pragma unroll
      for (int i = 0; i < 8; ++i) {
        float hv = mish_f(acc[i] + bias);
        atomicAdd(&gate_acc[i + 8 * g][gsel], hv * wv);  // ds_add_f32
      }
    }
  }
  __syncthreads();

  // ---- phase 4: band select + gates + tiny monotonic MLPs -----------------
  if (tid < 16) {
    const int row = tid;
    const int b = rowbase + row;
    if (b < B) {
      const float xs = xtile[row * 18 + 15];
      const float xk = xtile[row * 18 + 16];
      int band = (int)xtile[row * 18 + 17];
      band = band < 0 ? 0 : (band > 8 ? 8 : band);
      float result = heads_acc[row][band] + hb2[band];

      float msun = sun_b2[0], mstorm = storm_b2[0];
#pragma unroll
      for (int j = 0; j < 8; ++j) {
        msun   += tanh_f(xs * softplus_f(sun_w1[j])   + sun_b1[j])   * softplus_f(sun_w2[j]);
        mstorm += tanh_f(xk * softplus_f(storm_w1[j]) + storm_b1[j]) * softplus_f(storm_w2[j]);
      }
      result += sigmoid_f(gate_acc[row][0] + sb2[0]) * msun;
      result += sigmoid_f(gate_acc[row][1] + stb2[0]) * mstorm;
      out[b] = result;
    }
  }
}

extern "C" void kernel_launch(void* const* d_in, const int* in_sizes, int n_in,
                              void* d_out, int out_size, void* d_ws, size_t ws_size,
                              hipStream_t stream) {
  const float* x      = (const float*)d_in[0];
  const float* tw1    = (const float*)d_in[1];
  const float* tb1    = (const float*)d_in[2];
  const float* tw2    = (const float*)d_in[3];
  const float* tb2    = (const float*)d_in[4];
  const float* hw1    = (const float*)d_in[5];
  const float* hb1    = (const float*)d_in[6];
  const float* hw2    = (const float*)d_in[7];
  const float* hb2    = (const float*)d_in[8];
  const float* sw1    = (const float*)d_in[9];
  const float* sb1    = (const float*)d_in[10];
  const float* sw2    = (const float*)d_in[11];
  const float* sb2    = (const float*)d_in[12];
  const float* stw1   = (const float*)d_in[13];
  const float* stb1   = (const float*)d_in[14];
  const float* stw2   = (const float*)d_in[15];
  const float* stb2   = (const float*)d_in[16];
  const float* sun_w1 = (const float*)d_in[17];
  const float* sun_b1 = (const float*)d_in[18];
  const float* sun_w2 = (const float*)d_in[19];
  const float* sun_b2 = (const float*)d_in[20];
  const float* storm_w1 = (const float*)d_in[21];
  const float* storm_b1 = (const float*)d_in[22];
  const float* storm_w2 = (const float*)d_in[23];
  const float* storm_b2 = (const float*)d_in[24];
  (void)n_in; (void)ws_size;

  const int B = in_sizes[0] / 18;

  // Workspace layout (bf16 fragments): 32 + 256 + 576 + 32 + 32 frags * 512
  __bf16* W1 = (__bf16*)d_ws;
  __bf16* W2 = W1 + (size_t)32  * 512;
  __bf16* WH = W2 + (size_t)256 * 512;
  __bf16* WS = WH + (size_t)576 * 512;
  __bf16* WT = WS + (size_t)32  * 512;

  pack_wmma_b<<<64,  256, 0, stream>>>(tw1, W1, 15,  512, 1);
  pack_wmma_b<<<256, 256, 0, stream>>>(tw2, W2, 512, 256, 1);
  pack_wmma_b<<<512, 256, 0, stream>>>(hw1, WH, 256, 128, 9);
  pack_wmma_b<<<64,  256, 0, stream>>>(sw1, WS, 256, 64,  1);
  pack_wmma_b<<<64,  256, 0, stream>>>(stw1, WT, 256, 64, 1);

  const int nblocks = (B + 15) / 16;
  ionis_fused<<<nblocks, 256, 0, stream>>>(
      x, W1, W2, WH, WS, WT,
      tb1, tb2, hb1, hb2, hw2, sb1, sb2, sw2, stb1, stb2, stw2,
      sun_w1, sun_b1, sun_w2, sun_b2, storm_w1, storm_b1, storm_w2, storm_b2,
      (float*)d_out, B);
  (void)out_size;
}
